// SketchRNN_49349174231292
// MI455X (gfx1250) — matrix operand order repaired and checked by reading, not compiled, not run on hardware
//
#include <hip/hip_runtime.h>
#include <hip/hip_bf16.h>
#include <math.h>

// ---------------------------------------------------------------------------
// SketchRNN VAE on MI455X (gfx1250, wave32, WMMA f32_16x16x32_f16).
// Problem dims: B=256, T=250, Z=128, H=256, D=512, K=20.
// ---------------------------------------------------------------------------

typedef __attribute__((ext_vector_type(16))) _Float16 v16h;
typedef __attribute__((ext_vector_type(8)))  _Float16 v8h;
typedef __attribute__((ext_vector_type(8)))  float    v8f;

#define BATCH 256
#define TSTEPS 250
#define ZDIM 128
#define HDIM 256
#define DDIM 512
#define KMIX 20
#define POUT (6*KMIX+3)   // 123

// ---------------------------------------------------------------------------
// Repack a row-major f32 weight matrix W[K][N] into f16 WMMA B-fragments.
// Fragment layout: frag[kt][ntile][lane][e] with
//   K = kt*32 + (lane>>4)*16 + e,  N = ntile*16 + (lane&15)
// so each lane's 16 halves (32 B) are contiguous -> global_load_b128 x2.
// ---------------------------------------------------------------------------
__global__ __launch_bounds__(256)
void make_frags(const float* __restrict__ W, _Float16* __restrict__ F,
                int Kdim, int Ndim) {
  int idx = blockIdx.x * 256 + threadIdx.x;
  if (idx >= Kdim * Ndim) return;
  int e    = idx & 15;
  int lane = (idx >> 4) & 31;
  int tile = idx >> 9;
  int NT   = Ndim >> 4;
  int nt   = tile % NT;
  int kt   = tile / NT;
  int kk   = kt * 32 + (lane >> 4) * 16 + e;
  int nn   = nt * 16 + (lane & 15);
  F[idx] = (_Float16)W[(size_t)kk * Ndim + nn];
}

__global__ __launch_bounds__(256)
void zero_u32(unsigned int* __restrict__ p, int nWords) {
  int i = blockIdx.x * 256 + threadIdx.x;
  if (i < nWords) p[i] = 0u;
}

__device__ __forceinline__ float sigmoidf(float x) {
  return 1.0f / (1.0f + expf(-x));
}

// ---------------------------------------------------------------------------
// Fused LSTM step: one wave32 workgroup computes a 16x16 tile of h_new.
//   gate_pre = h_prev(f16) @ Wh(f16 frags)  [WMMA, f32 acc]
//            + x_t @ Wx (K=5, VALU)
//            + bias  (encoder)  |  zg[b,:]  (decoder; holds z@Wx_z + dec_b)
// Then the elementwise LSTM cell update with f32 c kept in place.
// ---------------------------------------------------------------------------
template<int HD>
__global__ __launch_bounds__(32)
void lstm_step(const float* __restrict__ data, int t_data, int TP1,
               const float* __restrict__ Wx,     // [5][4*HD]
               const float* __restrict__ bias,   // [4*HD] (nullptr for decoder)
               const float* __restrict__ zg,     // [B][4*HD] (nullptr for encoder)
               const _Float16* __restrict__ wfrag, // B-fragments of Wh
               const _Float16* __restrict__ h16in, // [B][HD] f16
               _Float16* __restrict__ h16out,      // [B][HD] f16 (ping-pong)
               float* __restrict__ c,              // [B][HD] f32 in-place
               float* __restrict__ hf32)           // [B][HD] f32 (for consumers)
{
  constexpr int KT    = HD / 32;
  constexpr int NTall = (4 * HD) / 16;
  constexpr int NTh   = HD / 16;

  const int lane = threadIdx.x;
  const int b0   = blockIdx.x * 16;
  const int nt   = blockIdx.y;        // column tile in H space
  const int n0   = nt * 16;
  const int half = lane >> 4;
  const int nc   = lane & 15;

  v8f a0 = {}, a1 = {}, a2 = {}, a3 = {};

  // A fragment source row: lane handles matrix row M = lane&15.
  const _Float16* hrow = h16in + (size_t)(b0 + nc) * HD;

  for (int kt = 0; kt < KT; ++kt) {
    const int kb = kt * 32;
    // A k order used by this code (matches B frags): half 0 -> K {kb..kb+15},
    //                                          half 1 -> K {kb+16..kb+31}.
    v16h a = *(const v16h*)(hrow + kb + half * 16);

    const _Float16* wk = wfrag + ((size_t)kt * NTall + nt) * 512 + lane * 16;
    v16h bI = *(const v16h*)(wk + (size_t)(0 * NTh) * 512);
    v16h bF = *(const v16h*)(wk + (size_t)(1 * NTh) * 512);
    v16h bG = *(const v16h*)(wk + (size_t)(2 * NTh) * 512);
    v16h bO = *(const v16h*)(wk + (size_t)(3 * NTh) * 512);

    if (kt + 1 < KT) {  // pull next K-tile of fragments toward the WGP
      __builtin_prefetch(wfrag + ((size_t)(kt + 1) * NTall + nt) * 512 + lane * 16, 0, 3);
    }

    a0 = __builtin_amdgcn_wmma_f32_16x16x32_f16(false, a, false, bI, (short)0, a0, false, false);
    a1 = __builtin_amdgcn_wmma_f32_16x16x32_f16(false, a, false, bF, (short)0, a1, false, false);
    a2 = __builtin_amdgcn_wmma_f32_16x16x32_f16(false, a, false, bG, (short)0, a2, false, false);
    a3 = __builtin_amdgcn_wmma_f32_16x16x32_f16(false, a, false, bO, (short)0, a3, false, false);
  }

  v8f accs[4] = {a0, a1, a2, a3};

  const int col = n0 + nc;            // j in [0, HD)
  float wxc[4][5];
  #pragma unroll
  for (int g = 0; g < 4; ++g)
    #pragma unroll
    for (int f = 0; f < 5; ++f)
      wxc[g][f] = Wx[(size_t)f * (4 * HD) + g * HD + col];

  float bz[4] = {0.f, 0.f, 0.f, 0.f};
  if (zg == nullptr) {
    #pragma unroll
    for (int g = 0; g < 4; ++g) bz[g] = bias[g * HD + col];
  }

  // C/D layout: lane owns N = lane&15; VGPR r holds M = (lane>>4)*8 + r.
  #pragma unroll
  for (int r = 0; r < 8; ++r) {
    const int row = b0 + half * 8 + r;
    const float* xr = data + ((size_t)row * TP1 + t_data) * 5;
    const float x0 = xr[0], x1 = xr[1], x2 = xr[2], x3 = xr[3], x4 = xr[4];

    float pre[4];
    #pragma unroll
    for (int g = 0; g < 4; ++g) {
      float base = (zg != nullptr) ? zg[(size_t)row * (4 * HD) + g * HD + col]
                                   : bz[g];
      pre[g] = accs[g][r] + base
             + x0 * wxc[g][0] + x1 * wxc[g][1] + x2 * wxc[g][2]
             + x3 * wxc[g][3] + x4 * wxc[g][4];
    }

    const float ig = sigmoidf(pre[0]);
    const float fg = sigmoidf(pre[1]);
    const float gg = tanhf(pre[2]);
    const float og = sigmoidf(pre[3]);

    const size_t offe = (size_t)row * HD + col;
    const float cn = fg * c[offe] + ig * gg;
    const float hn = og * tanhf(cn);
    c[offe]     = cn;
    hf32[offe]  = hn;
    h16out[offe] = (_Float16)hn;
  }
}

// ---------------------------------------------------------------------------
// Encoder head: y = [h_f, h_b] @ enc_out_W + b ; split into (z_mean, z_logvar);
// reparameterize z = mean + exp(0.5*logvar) * eps. One block per batch row.
// ---------------------------------------------------------------------------
__global__ __launch_bounds__(256)
void enc_out_sample(const float* __restrict__ hf, const float* __restrict__ hb,
                    const float* __restrict__ W, const float* __restrict__ bb,
                    const float* __restrict__ eps,
                    float* __restrict__ zmean_out, float* __restrict__ zlv_out,
                    float* __restrict__ z) {
  __shared__ float hc[2 * HDIM];
  __shared__ float yr[2 * ZDIM];
  const int b = blockIdx.x, tid = threadIdx.x;
  hc[tid]        = hf[(size_t)b * HDIM + tid];
  hc[HDIM + tid] = hb[(size_t)b * HDIM + tid];
  __syncthreads();
  float acc = bb[tid];
  for (int k = 0; k < 2 * HDIM; ++k) acc += hc[k] * W[(size_t)k * (2 * ZDIM) + tid];
  yr[tid] = acc;
  __syncthreads();
  if (tid < ZDIM) {
    const float zm = yr[tid], zl = yr[ZDIM + tid];
    zmean_out[(size_t)b * ZDIM + tid] = zm;
    zlv_out[(size_t)b * ZDIM + tid]   = zl;
    z[(size_t)b * ZDIM + tid] = zm + expf(0.5f * zl) * eps[(size_t)b * ZDIM + tid];
  }
}

// ---------------------------------------------------------------------------
// Decoder init: [h0, c0] = tanh(z @ init_W + init_b). One block per batch row.
// ---------------------------------------------------------------------------
__global__ __launch_bounds__(256)
void dec_init(const float* __restrict__ z, const float* __restrict__ W,
              const float* __restrict__ bb,
              _Float16* __restrict__ h16, float* __restrict__ c) {
  __shared__ float zr[ZDIM];
  const int b = blockIdx.x, tid = threadIdx.x;
  if (tid < ZDIM) zr[tid] = z[(size_t)b * ZDIM + tid];
  __syncthreads();
  for (int cc = 0; cc < 4; ++cc) {
    const int col = tid + cc * 256;          // 0..1023
    float acc = bb[col];
    for (int k = 0; k < ZDIM; ++k) acc += zr[k] * W[(size_t)k * (2 * DDIM) + col];
    const float v = tanhf(acc);
    if (col < DDIM) h16[(size_t)b * DDIM + col] = (_Float16)v;
    else            c[(size_t)b * DDIM + col - DDIM] = v;
  }
}

// ---------------------------------------------------------------------------
// Decoder z-gate precompute: zg = z @ dec_Wx[5:, :] + dec_b  (B x 4D).
// ---------------------------------------------------------------------------
__global__ __launch_bounds__(256)
void dec_zgate(const float* __restrict__ z, const float* __restrict__ decWx,
               const float* __restrict__ db, float* __restrict__ zg) {
  __shared__ float zr[ZDIM];
  const int b = blockIdx.x, tid = threadIdx.x;
  if (tid < ZDIM) zr[tid] = z[(size_t)b * ZDIM + tid];
  __syncthreads();
  for (int cc = 0; cc < 8; ++cc) {
    const int col = tid + cc * 256;          // 0..2047
    float acc = db[col];
    for (int k = 0; k < ZDIM; ++k)
      acc += zr[k] * decWx[(size_t)(5 + k) * (4 * DDIM) + col];
    zg[(size_t)b * (4 * DDIM) + col] = acc;
  }
}

// ---------------------------------------------------------------------------
// Per-timestep mixture head: y = h_t @ mix_W + mix_b, then
// softmax(pi) | mu | exp(sigma) | tanh(rho) | pen -> params[:, t, :].
// One block per batch row.
// ---------------------------------------------------------------------------
__global__ __launch_bounds__(128)
void mix_head(const float* __restrict__ h, const float* __restrict__ mixW,
              const float* __restrict__ mixb, float* __restrict__ outp, int t) {
  __shared__ float hrow[DDIM];
  __shared__ float yrow[128];
  __shared__ float red[2];
  const int b = blockIdx.x, tid = threadIdx.x;
  for (int i = tid; i < DDIM; i += 128) hrow[i] = h[(size_t)b * DDIM + i];
  __syncthreads();
  if (tid < POUT) {
    float acc = mixb[tid];
    for (int k = 0; k < DDIM; ++k) acc += hrow[k] * mixW[(size_t)k * POUT + tid];
    yrow[tid] = acc;
  }
  __syncthreads();
  if (tid == 0) {
    float m = yrow[0];
    for (int i = 1; i < KMIX; ++i) m = fmaxf(m, yrow[i]);
    float s = 0.f;
    for (int i = 0; i < KMIX; ++i) s += expf(yrow[i] - m);
    red[0] = m; red[1] = s;
  }
  __syncthreads();
  if (tid < POUT) {
    const float y = yrow[tid];
    float v;
    if      (tid < KMIX)      v = expf(y - red[0]) / red[1]; // pi
    else if (tid < 3 * KMIX)  v = y;                          // mu
    else if (tid < 5 * KMIX)  v = expf(y);                    // sigma
    else if (tid < 6 * KMIX)  v = tanhf(y);                   // rho
    else                      v = y;                          // pen logits
    outp[((size_t)b * TSTEPS + t) * POUT + tid] = v;
  }
}

// ---------------------------------------------------------------------------
// Host-side orchestration (graph-capturable: only kernel launches on stream).
// ---------------------------------------------------------------------------
extern "C" void kernel_launch(void* const* d_in, const int* in_sizes, int n_in,
                              void* d_out, int out_size, void* d_ws, size_t ws_size,
                              hipStream_t stream) {
  const float* data      = (const float*)d_in[0];   // (B, T+1, 5)
  const float* eps       = (const float*)d_in[1];   // (B, Z)
  const float* enc_Wx_f  = (const float*)d_in[2];   // (5, 4H)
  const float* enc_Wh_f  = (const float*)d_in[3];   // (H, 4H)
  const float* enc_b_f   = (const float*)d_in[4];   // (4H,)
  const float* enc_Wx_b  = (const float*)d_in[5];
  const float* enc_Wh_b  = (const float*)d_in[6];
  const float* enc_b_b   = (const float*)d_in[7];
  const float* enc_out_W = (const float*)d_in[8];   // (2H, 2Z)
  const float* enc_out_b = (const float*)d_in[9];
  const float* init_W    = (const float*)d_in[10];  // (Z, 2D)
  const float* init_b    = (const float*)d_in[11];
  const float* dec_Wx    = (const float*)d_in[12];  // (5+Z, 4D)
  const float* dec_Wh    = (const float*)d_in[13];  // (D, 4D)
  const float* dec_b     = (const float*)d_in[14];
  const float* mix_W     = (const float*)d_in[15];  // (D, 6K+3)
  const float* mix_b     = (const float*)d_in[16];

  float* outp      = (float*)d_out;                       // params (B,T,123)
  float* zmean_out = outp + (size_t)BATCH * TSTEPS * POUT;
  float* zlv_out   = zmean_out + (size_t)BATCH * ZDIM;

  // ---- workspace carve-out (~9 MB) ----
  char* w = (char*)d_ws;
  size_t off = 0;
  auto carve = [&](size_t bytes) -> char* {
    char* p = w + off;
    off += (bytes + 255) & ~(size_t)255;
    return p;
  };
  _Float16* wfe_f = (_Float16*)carve((size_t)HDIM * 4 * HDIM * 2); // 512 KB
  _Float16* wfe_b = (_Float16*)carve((size_t)HDIM * 4 * HDIM * 2);
  _Float16* wfd   = (_Float16*)carve((size_t)DDIM * 4 * DDIM * 2); // 2 MB
  _Float16* hf16[2], *hb16[2], *hd16[2];
  hf16[0] = (_Float16*)carve((size_t)BATCH * HDIM * 2);
  hf16[1] = (_Float16*)carve((size_t)BATCH * HDIM * 2);
  hb16[0] = (_Float16*)carve((size_t)BATCH * HDIM * 2);
  hb16[1] = (_Float16*)carve((size_t)BATCH * HDIM * 2);
  hd16[0] = (_Float16*)carve((size_t)BATCH * DDIM * 2);
  hd16[1] = (_Float16*)carve((size_t)BATCH * DDIM * 2);
  float* cf   = (float*)carve((size_t)BATCH * HDIM * 4);
  float* cb   = (float*)carve((size_t)BATCH * HDIM * 4);
  float* cd   = (float*)carve((size_t)BATCH * DDIM * 4);
  float* hff  = (float*)carve((size_t)BATCH * HDIM * 4);   // f32 h (fwd enc)
  float* hbf  = (float*)carve((size_t)BATCH * HDIM * 4);   // f32 h (bwd enc)
  float* hdf  = (float*)carve((size_t)BATCH * DDIM * 4);   // f32 h (dec)
  float* zg   = (float*)carve((size_t)BATCH * 4 * DDIM * 4); // 2 MB
  float* z    = (float*)carve((size_t)BATCH * ZDIM * 4);
  (void)ws_size; (void)in_sizes; (void)n_in; (void)out_size;

  const int TP1 = TSTEPS + 1;

  // ---- one-time prep (re-done every call; ws is not persistent) ----
  make_frags<<<(HDIM * 4 * HDIM) / 256, 256, 0, stream>>>(enc_Wh_f, wfe_f, HDIM, 4 * HDIM);
  make_frags<<<(HDIM * 4 * HDIM) / 256, 256, 0, stream>>>(enc_Wh_b, wfe_b, HDIM, 4 * HDIM);
  make_frags<<<(DDIM * 4 * DDIM) / 256, 256, 0, stream>>>(dec_Wh,   wfd,   DDIM, 4 * DDIM);

  const int hWords = (BATCH * HDIM * 2) / 4;  // f16 buffer in u32 words
  const int cWords = (BATCH * HDIM * 4) / 4;
  zero_u32<<<(hWords + 255) / 256, 256, 0, stream>>>((unsigned int*)hf16[0], hWords);
  zero_u32<<<(hWords + 255) / 256, 256, 0, stream>>>((unsigned int*)hb16[0], hWords);
  zero_u32<<<(cWords + 255) / 256, 256, 0, stream>>>((unsigned int*)cf, cWords);
  zero_u32<<<(cWords + 255) / 256, 256, 0, stream>>>((unsigned int*)cb, cWords);

  // ---- bidirectional encoder: 250 fwd + 250 bwd steps ----
  const dim3 gridE(BATCH / 16, HDIM / 16);  // (16,16) tiles, 1 wave each
  for (int s = 0; s < TSTEPS; ++s) {
    const int pi = s & 1, po = 1 - pi;
    // forward direction reads enc_in[s] = data[:, 1+s, :]
    lstm_step<HDIM><<<gridE, 32, 0, stream>>>(
        data, 1 + s, TP1, enc_Wx_f, enc_b_f, nullptr,
        wfe_f, hf16[pi], hf16[po], cf, hff);
    // backward direction reads enc_in[T-1-s] = data[:, T-s, :]
    lstm_step<HDIM><<<gridE, 32, 0, stream>>>(
        data, TSTEPS - s, TP1, enc_Wx_b, enc_b_b, nullptr,
        wfe_b, hb16[pi], hb16[po], cb, hbf);
  }

  // ---- encoder head + reparameterization ----
  enc_out_sample<<<BATCH, 256, 0, stream>>>(hff, hbf, enc_out_W, enc_out_b,
                                            eps, zmean_out, zlv_out, z);

  // ---- decoder init state + z-gate precompute ----
  dec_init<<<BATCH, 256, 0, stream>>>(z, init_W, init_b, hd16[0], cd);
  dec_zgate<<<BATCH, 256, 0, stream>>>(z, dec_Wx, dec_b, zg);

  // ---- decoder: 250 steps, each followed by the fused mixture head ----
  const dim3 gridD(BATCH / 16, DDIM / 16);  // (16,32) tiles
  for (int s = 0; s < TSTEPS; ++s) {
    const int pi = s & 1, po = 1 - pi;
    // decoder input is dec_in[s] = data[:, s, :]
    lstm_step<DDIM><<<gridD, 32, 0, stream>>>(
        data, s, TP1, dec_Wx, nullptr, zg,
        wfd, hd16[pi], hd16[po], cd, hdf);
    mix_head<<<BATCH, 128, 0, stream>>>(hdf, mix_W, mix_b, outp, s);
  }
}
